// NeuralMachineTranslator_73624329388144
// MI455X (gfx1250) — matrix operand order repaired and checked by reading, not compiled, not run on hardware
//
#include <hip/hip_runtime.h>
#include <hip/hip_bf16.h>
#include <math.h>

// ---------------------------------------------------------------------------
// Types for CDNA5 WMMA (wave32): 16x16x32 bf16 -> f32
// ---------------------------------------------------------------------------
typedef __attribute__((ext_vector_type(16))) __bf16        v16bf;
typedef __attribute__((ext_vector_type(8)))  __bf16        v8bf;
typedef __attribute__((ext_vector_type(8)))  float         v8f;
typedef __attribute__((ext_vector_type(4)))  unsigned int  v4u;
typedef __attribute__((ext_vector_type(8)))  unsigned int  v8u;

static __device__ __forceinline__ unsigned short f32_to_bf16_rne(float f) {
    unsigned int u = __float_as_uint(f);
    unsigned int rounding = 0x7FFFu + ((u >> 16) & 1u);
    return (unsigned short)((u + rounding) >> 16);
}

static __device__ __forceinline__ float sigmoidf(float x) {
    return 1.0f / (1.0f + __expf(-x));
}

// ---------------------------------------------------------------------------
// TDM issue: DMA a 64-row x 32-col bf16 tile of W (N x K row-major) into LDS.
// D# built per CDNA5 ISA 8.3/8.4: group0 {count=1, lds_addr, global_addr,
// type=2}, group1 {data_size=2B, tensor_dim0=K, tensor_dim1=64 rows,
// tile_dim0=32, tile_dim1=64, tensor_dim0_stride=K}. 2D tensor -> 2 SGPR
// groups (VADDR2/3 = NULL form). Tracked by TENSORcnt.
// ---------------------------------------------------------------------------
static __device__ __forceinline__ void tdm_load_w_tile(const unsigned short* W,
                                                       unsigned row0, unsigned k0,
                                                       unsigned K, unsigned lds_byte) {
    unsigned long long gaddr = (unsigned long long)(uintptr_t)W +
                               ((unsigned long long)row0 * K + k0) * 2ull;
    v4u g0;
    g0.x = 1u;                                              // count=1 (valid), no gather
    g0.y = lds_byte;                                        // LDS byte address
    g0.z = (unsigned)(gaddr & 0xFFFFFFFFu);                 // global_addr[31:0]
    g0.w = (unsigned)((gaddr >> 32) & 0x01FFFFFFu) | (2u << 30); // [56:32] | type=2
    v8u g1;
    g1[0] = (1u << 16);                                     // data_size=1 -> 2 bytes
    g1[1] = (K & 0xFFFFu) << 16;                            // tensor_dim0[15:0]
    g1[2] = (K >> 16) | (64u << 16);                        // tensor_dim0[31:16] | tensor_dim1 lo
    g1[3] = (32u << 16);                                    // tensor_dim1 hi(0) | tile_dim0=32
    g1[4] = 64u;                                            // tile_dim1=64, tile_dim2=0
    g1[5] = K;                                              // tensor_dim0_stride[31:0]
    g1[6] = 0u;                                             // stride hi | dim1_stride lo
    g1[7] = 0u;
    asm volatile("tensor_load_to_lds %0, %1" : : "s"(g0), "s"(g1) : "memory");
}

// ---------------------------------------------------------------------------
// f32 -> bf16 conversion (weights, once per call)
// ---------------------------------------------------------------------------
__global__ void nmt_cvt_bf16_kernel(const float* __restrict__ in,
                                    unsigned short* __restrict__ out, int n) {
    int i = blockIdx.x * blockDim.x + threadIdx.x;
    if (i < n) out[i] = f32_to_bf16_rne(in[i]);
}

// ---------------------------------------------------------------------------
// Encoder: enc[b][s][0:E] = enc_emb[src[b][s]], enc[b][s][E:2E] = pos_emb[s+1]
// ---------------------------------------------------------------------------
__global__ void nmt_encoder_kernel(const int* __restrict__ src,
                                   const float* __restrict__ enc_emb,
                                   const float* __restrict__ pos_emb,
                                   float* __restrict__ enc,
                                   int B, int S, int E) {
    int H = 2 * E;
    int idx = blockIdx.x * blockDim.x + threadIdx.x;
    if (idx >= B * S * H) return;
    int h  = idx % H;
    int bs = idx / H;
    int s  = bs % S;
    int b  = bs / S;
    float v;
    if (h < E) v = enc_emb[(size_t)src[b * S + s] * E + h];
    else       v = pos_emb[(size_t)(s + 1) * E + (h - E)];
    enc[idx] = v;
}

// h0 = enc.sum(axis=1)/S ; c = context0
__global__ void nmt_init_state_kernel(const float* __restrict__ enc,
                                      const float* __restrict__ context0,
                                      float* __restrict__ h, float* __restrict__ c,
                                      int B, int S, int H) {
    int idx = blockIdx.x * blockDim.x + threadIdx.x;
    if (idx >= B * H) return;
    int hh = idx % H;
    int b  = idx / H;
    float sum = 0.f;
    for (int s = 0; s < S; ++s) sum += enc[((size_t)(b * S + s)) * H + hh];
    h[idx] = sum / (float)S;
    c[idx] = context0[idx];
}

__global__ void nmt_zero_loss_kernel(float* loss) { if (threadIdx.x == 0) *loss = 0.f; }

// ---------------------------------------------------------------------------
// Attention scores: scores[b][s] = dot(enc[b][s], h[b]) over H. One wave each.
// ---------------------------------------------------------------------------
__global__ void nmt_attn_scores_kernel(const float* __restrict__ enc,
                                       const float* __restrict__ h,
                                       float* __restrict__ scores,
                                       int S, int H) {
    int bs   = blockIdx.x;           // b*S + s
    int lane = threadIdx.x & 31;
    int b    = bs / S;
    const float* e  = enc + (size_t)bs * H;
    const float* hb = h   + (size_t)b  * H;
    float sum = 0.f;
    for (int k = lane; k < H; k += 32) sum += e[k] * hb[k];
    #pragma unroll
    for (int off = 16; off > 0; off >>= 1) sum += __shfl_down(sum, off, 32);
    if (lane == 0) scores[bs] = sum;
}

// softmax over S per batch row (one wave per b)
__global__ void nmt_softmax_kernel(const float* __restrict__ scores,
                                   float* __restrict__ a, int S) {
    int b = blockIdx.x;
    int lane = threadIdx.x & 31;
    const float* row = scores + (size_t)b * S;
    float m = -3.0e38f;
    for (int s = lane; s < S; s += 32) m = fmaxf(m, row[s]);
    #pragma unroll
    for (int off = 16; off > 0; off >>= 1) m = fmaxf(m, __shfl_xor(m, off, 32));
    float sum = 0.f;
    for (int s = lane; s < S; s += 32) sum += __expf(row[s] - m);
    #pragma unroll
    for (int off = 16; off > 0; off >>= 1) sum += __shfl_xor(sum, off, 32);
    float inv = 1.0f / sum;
    for (int s = lane; s < S; s += 32) a[(size_t)b * S + s] = __expf(row[s] - m) * inv;
}

// ws[b][hh] = sum_s a[b][s]*enc[b][s][hh]; cat = [ws, h] as bf16 (B x 2H)
__global__ void nmt_wsum_cat_kernel(const float* __restrict__ a,
                                    const float* __restrict__ enc,
                                    const float* __restrict__ h,
                                    unsigned short* __restrict__ cat,
                                    int B, int S, int H) {
    int idx = blockIdx.x * blockDim.x + threadIdx.x;
    if (idx >= B * H) return;
    int hh = idx % H;
    int b  = idx / H;
    float acc = 0.f;
    const float* arow = a + (size_t)b * S;
    for (int s = 0; s < S; ++s) acc += arow[s] * enc[((size_t)(b * S + s)) * H + hh];
    cat[(size_t)b * 2 * H + hh]     = f32_to_bf16_rne(acc);
    cat[(size_t)b * 2 * H + H + hh] = f32_to_bf16_rne(h[idx]);
}

// ---------------------------------------------------------------------------
// WMMA GEMM with TDM-staged weights:
//   C(64 x N, f32) = A(64 x K, bf16) * W(N x K, bf16)^T + bias
// Block = 4 waves (128 threads); wave w owns M-tile w; block covers a
// 64(M) x 64(N) super-tile. Per K-step of 32:
//   - TDM DMAs the shared 64x32 W tile into LDS (double buffered, TENSORcnt)
//   - each wave loads its A fragment from global (2x16B) and 4 B fragments
//     from LDS (2x ds_load_b128 each), issuing 4 back-to-back WMMAs.
// Requires M == 64, N % 64 == 0, K % 32 == 0. grid = dim3(N/64), block = 128.
// ---------------------------------------------------------------------------
__global__ void nmt_gemm_bf16_wmma4_tdm(const unsigned short* __restrict__ A,
                                        const unsigned short* __restrict__ W,
                                        const float* __restrict__ bias,
                                        float* __restrict__ C,
                                        unsigned short* __restrict__ Cbf,
                                        int M, int N, int K) {
    __shared__ __align__(64) unsigned short sW[2][64 * 32];  // 2 x 4 KB tiles
    int lane   = threadIdx.x & 31;
    int wave   = threadIdx.x >> 5;               // 0..3 -> M tile (M = 64)
    int mtile  = wave;
    int ntile0 = blockIdx.x * 4;                 // 4 consecutive N tiles
    int l15    = lane & 15;
    int khalf  = lane >> 4;                      // 0 or 1

    const unsigned short* arow = A + (size_t)(mtile * 16 + l15) * K;
    unsigned row0 = (unsigned)(ntile0 * 16);
    unsigned lds0 = (unsigned)(uintptr_t)&sW[0][0];
    unsigned lds1 = (unsigned)(uintptr_t)&sW[1][0];
    int NT = K >> 5;

    if (wave == 0) tdm_load_w_tile(W, row0, 0u, (unsigned)K, lds0);

    v8f acc0 = {}, acc1 = {}, acc2 = {}, acc3 = {};
    int cur = 0;
    for (int kt = 0; kt < NT; ++kt) {
        if (wave == 0) __builtin_amdgcn_s_wait_tensorcnt(0);
        __syncthreads();                                   // buf[cur] ready for all waves
        if (wave == 0 && (kt + 1) < NT)                    // prefetch next tile into other buffer
            tdm_load_w_tile(W, row0, (unsigned)((kt + 1) << 5), (unsigned)K,
                            cur ? lds0 : lds1);

        int k0 = kt << 5;
        int ka = k0 + khalf * 16;
        union { v16bf v; v8bf h[2]; } au;
        au.h[0] = *(const v8bf*)(arow + ka);               // K = ka .. ka+7
        au.h[1] = *(const v8bf*)(arow + ka + 8);           // K = ka+8 .. ka+15

        const unsigned short* lw = &sW[cur][0];            // 64 rows x 32 cols
        int kb = khalf * 16;
        v16bf b0 = *(const v16bf*)(lw + (0 * 16 + l15) * 32 + kb);
        v16bf b1 = *(const v16bf*)(lw + (1 * 16 + l15) * 32 + kb);
        v16bf b2 = *(const v16bf*)(lw + (2 * 16 + l15) * 32 + kb);
        v16bf b3 = *(const v16bf*)(lw + (3 * 16 + l15) * 32 + kb);

        acc0 = __builtin_amdgcn_wmma_f32_16x16x32_bf16(false, au.v, false, b0, (short)0, acc0, false, false);
        acc1 = __builtin_amdgcn_wmma_f32_16x16x32_bf16(false, au.v, false, b1, (short)0, acc1, false, false);
        acc2 = __builtin_amdgcn_wmma_f32_16x16x32_bf16(false, au.v, false, b2, (short)0, acc2, false, false);
        acc3 = __builtin_amdgcn_wmma_f32_16x16x32_bf16(false, au.v, false, b3, (short)0, acc3, false, false);
        cur ^= 1;
    }

    int rowa = mtile * 16 + khalf * 8;           // lanes 16-31 hold M rows 8..15 of tile
    #pragma unroll
    for (int j = 0; j < 4; ++j) {
        const v8f* accp = (j == 0) ? &acc0 : (j == 1) ? &acc1 : (j == 2) ? &acc2 : &acc3;
        int col = (ntile0 + j) * 16 + l15;
        float bv = bias ? bias[col] : 0.f;
        #pragma unroll
        for (int r = 0; r < 8; ++r) {
            float v = (*accp)[r] + bv;
            size_t idx = (size_t)(rowa + r) * N + col;
            if (C)   C[idx]   = v;
            if (Cbf) Cbf[idx] = f32_to_bf16_rne(v);
        }
    }
}

// ---------------------------------------------------------------------------
// LSTM pointwise + small x@W_ih^T (K=100) term folded in.
// ---------------------------------------------------------------------------
__global__ void nmt_lstm_kernel(const float* __restrict__ gates,   // B x 4H (h_att @ W_hh^T)
                                const int* __restrict__ trg, int t, int T,
                                const float* __restrict__ dec_emb,
                                const float* __restrict__ W_ih,
                                const float* __restrict__ b_ih,
                                const float* __restrict__ b_hh,
                                float* __restrict__ h, float* __restrict__ c,
                                unsigned short* __restrict__ hbf,
                                int B, int H, int D) {
    int idx = blockIdx.x * blockDim.x + threadIdx.x;
    if (idx >= B * H) return;
    int j = idx % H;
    int b = idx / H;
    int y = trg[b * T + t];
    const float* x = dec_emb + (size_t)y * D;
    float g4[4];
    #pragma unroll
    for (int q = 0; q < 4; ++q) {
        int row = q * H + j;
        float s = gates[(size_t)b * 4 * H + row] + b_ih[row] + b_hh[row];
        const float* wr = W_ih + (size_t)row * D;
        for (int k = 0; k < D; ++k) s += x[k] * wr[k];
        g4[q] = s;
    }
    float ig = sigmoidf(g4[0]);
    float fg = sigmoidf(g4[1]);
    float gg = tanhf(g4[2]);
    float og = sigmoidf(g4[3]);
    float cn = fg * c[idx] + ig * gg;
    float hn = og * tanhf(cn);
    c[idx] = cn;
    h[idx] = hn;
    hbf[idx] = f32_to_bf16_rne(hn);
}

// ---------------------------------------------------------------------------
// log-softmax NLL + argmax over V per batch row; accumulate loss, write pred.
// ---------------------------------------------------------------------------
__global__ void nmt_loss_kernel(const float* __restrict__ logits,
                                const int* __restrict__ trg,
                                const int* __restrict__ trg_lengths,
                                int t, int T, int V, int B,
                                float* __restrict__ preds_out,
                                float* __restrict__ loss_out) {
    __shared__ float sv[256];
    __shared__ int   si[256];
    int b   = blockIdx.x;
    int tid = threadIdx.x;
    const float* row = logits + (size_t)b * V;

    float m = -3.0e38f; int mi = 0;
    for (int v = tid; v < V; v += 256) {
        float x = row[v];
        if (x > m) { m = x; mi = v; }
    }
    sv[tid] = m; si[tid] = mi;
    __syncthreads();
    for (int s = 128; s > 0; s >>= 1) {
        if (tid < s) {
            if (sv[tid + s] > sv[tid] ||
                (sv[tid + s] == sv[tid] && si[tid + s] < si[tid])) {
                sv[tid] = sv[tid + s]; si[tid] = si[tid + s];
            }
        }
        __syncthreads();
    }
    float rmax = sv[0];
    int   amax = si[0];
    __syncthreads();

    float sum = 0.f;
    for (int v = tid; v < V; v += 256) sum += __expf(row[v] - rmax);
    sv[tid] = sum;
    __syncthreads();
    for (int s = 128; s > 0; s >>= 1) {
        if (tid < s) sv[tid] += sv[tid + s];
        __syncthreads();
    }

    if (tid == 0) {
        float lse = rmax + __logf(sv[0]);
        int y = trg[b * T + t];
        float nll = lse - row[y];
        preds_out[b * T + t] = (float)amax;
        if (t < trg_lengths[b]) atomicAdd(loss_out, nll / (float)B);
    }
}

// ---------------------------------------------------------------------------
// Host-side orchestration
// ---------------------------------------------------------------------------
extern "C" void kernel_launch(void* const* d_in, const int* in_sizes, int n_in,
                              void* d_out, int out_size, void* d_ws, size_t ws_size,
                              hipStream_t stream) {
    const int B = 64, S = 50, T = 50, E = 512, V = 32000, D = 100;
    const int H = 2 * E;          // 1024
    const int H2 = 2 * H;         // 2048
    const int H4 = 4 * H;         // 4096

    const int*   src      = (const int*)  d_in[0];
    const int*   trg      = (const int*)  d_in[1];
    const int*   trg_len  = (const int*)  d_in[2];
    const float* enc_emb  = (const float*)d_in[3];
    const float* pos_emb  = (const float*)d_in[4];
    const float* attn_W   = (const float*)d_in[5];
    const float* attn_b   = (const float*)d_in[6];
    const float* dec_emb  = (const float*)d_in[7];
    const float* W_ih     = (const float*)d_in[8];
    const float* W_hh     = (const float*)d_in[9];
    const float* b_ih     = (const float*)d_in[10];
    const float* b_hh     = (const float*)d_in[11];
    const float* out_W    = (const float*)d_in[12];
    const float* out_b    = (const float*)d_in[13];
    const float* context0 = (const float*)d_in[14];

    // workspace carve
    char* w = (char*)d_ws;
    auto carve = [&](size_t bytes) -> char* {
        char* p = w;
        w += (bytes + 255) & ~(size_t)255;
        return p;
    };
    float*          enc     = (float*)         carve((size_t)B * S * H * 4);  // 13.1 MB
    unsigned short* attnWb  = (unsigned short*)carve((size_t)H * H2 * 2);     //  4.2 MB
    unsigned short* Whhb    = (unsigned short*)carve((size_t)H4 * H * 2);     //  8.4 MB
    unsigned short* outWb   = (unsigned short*)carve((size_t)V * H * 2);      // 65.5 MB (L2-resident)
    float*          hstate  = (float*)         carve((size_t)B * H * 4);
    float*          cstate  = (float*)         carve((size_t)B * H * 4);
    unsigned short* hbf     = (unsigned short*)carve((size_t)B * H * 2);
    float*          scores  = (float*)         carve((size_t)B * S * 4);
    float*          attw    = (float*)         carve((size_t)B * S * 4);
    unsigned short* catb    = (unsigned short*)carve((size_t)B * H2 * 2);
    unsigned short* hattb   = (unsigned short*)carve((size_t)B * H * 2);
    float*          gates   = (float*)         carve((size_t)B * H4 * 4);
    float*          logits  = (float*)         carve((size_t)B * V * 4);      //  8.2 MB

    float* preds_out = (float*)d_out;          // B*T preds (as float)
    float* loss_out  = preds_out + B * T;      // scalar loss

    // one-time (per call) weight conversion to bf16
    {
        int n;
        n = H * H2;  nmt_cvt_bf16_kernel<<<(n + 255) / 256, 256, 0, stream>>>(attn_W, attnWb, n);
        n = H4 * H;  nmt_cvt_bf16_kernel<<<(n + 255) / 256, 256, 0, stream>>>(W_hh, Whhb, n);
        n = V * H;   nmt_cvt_bf16_kernel<<<(n + 255) / 256, 256, 0, stream>>>(out_W, outWb, n);
    }

    // encoder + initial state + zero loss
    {
        int n = B * S * H;
        nmt_encoder_kernel<<<(n + 255) / 256, 256, 0, stream>>>(src, enc_emb, pos_emb, enc, B, S, E);
        n = B * H;
        nmt_init_state_kernel<<<(n + 255) / 256, 256, 0, stream>>>(enc, context0, hstate, cstate, B, S, H);
        nmt_zero_loss_kernel<<<1, 32, 0, stream>>>(loss_out);
    }

    // 50 sequential decoder steps
    for (int t = 0; t < T; ++t) {
        nmt_attn_scores_kernel<<<B * S, 32, 0, stream>>>(enc, hstate, scores, S, H);
        nmt_softmax_kernel<<<B, 32, 0, stream>>>(scores, attw, S);
        {
            int n = B * H;
            nmt_wsum_cat_kernel<<<(n + 255) / 256, 256, 0, stream>>>(attw, enc, hstate, catb, B, S, H);
        }
        // h_att = cat @ attn_W^T + attn_b  (M=64, N=1024, K=2048) -> bf16 only
        nmt_gemm_bf16_wmma4_tdm<<<dim3(H / 64), 128, 0, stream>>>(
            catb, attnWb, attn_b, nullptr, hattb, B, H, H2);
        // gates_hh = h_att @ W_hh^T  (M=64, N=4096, K=1024) -> f32
        nmt_gemm_bf16_wmma4_tdm<<<dim3(H4 / 64), 128, 0, stream>>>(
            hattb, Whhb, nullptr, gates, nullptr, B, H4, H);
        // LSTM pointwise (+ x@W_ih^T, K=100, scalar)
        {
            int n = B * H;
            nmt_lstm_kernel<<<(n + 255) / 256, 256, 0, stream>>>(
                gates, trg, t, T, dec_emb, W_ih, b_ih, b_hh, hstate, cstate, hbf, B, H, D);
        }
        // logits = h_new @ out_W^T + out_b  (M=64, N=32000, K=1024) -> f32
        nmt_gemm_bf16_wmma4_tdm<<<dim3(V / 64), 128, 0, stream>>>(
            hbf, outWb, out_b, logits, nullptr, B, V, H);
        // log-softmax NLL + argmax
        nmt_loss_kernel<<<B, 256, 0, stream>>>(
            logits, trg, trg_len, t, T, V, B, preds_out, loss_out);
    }
}
